// TopKMoELayer_39779987096107
// MI455X (gfx1250) — compile-verified, matrix operand order checked by reading, not run on hardware
//
#include <hip/hip_runtime.h>
#include <hip/hip_bf16.h>

// ---------------------------------------------------------------------------
// Top-2 MoE forward for MI455X (gfx1250): bf16 WMMA + Tensor Data Mover.
//   B=4, S=2048, H=1024, E=8, K=2, I=2048, T=B*S=8192
// Output: [T*H] fp32 MoE output, then 1 fp32 aux loss.
// ---------------------------------------------------------------------------

#define TT 8192     // tokens
#define HH 1024     // hidden
#define EE 8        // experts
#define II 2048     // intermediate
#define TM 16       // token tile per expert block
#define ICH 256     // I-chunk staged in LDS

typedef __attribute__((ext_vector_type(16))) __bf16 v16bf;
typedef __attribute__((ext_vector_type(8)))  __bf16 v8bf;
typedef __attribute__((ext_vector_type(8)))  float  v8f;
typedef __attribute__((ext_vector_type(4)))  unsigned int v4u;
typedef __attribute__((ext_vector_type(8)))  int v8i;
typedef __attribute__((ext_vector_type(4)))  int v4i;

__device__ __forceinline__ v8f wmma_bf16(v16bf a, v16bf b, v8f c) {
  return __builtin_amdgcn_wmma_f32_16x16x32_bf16(false, a, false, b, (short)0, c,
                                                 false, false);
}

// --------------------------- TDM descriptor helpers ------------------------
// 2-D tile load: tile_dim0 = tk elements along contiguous axis, tile_dim1 = tr
// rows, row stride = `stride` elements, element size 2 bytes (bf16).
__device__ __forceinline__ void tdm_load_2d(unsigned lds_addr, const void* gptr,
                                            unsigned tk, unsigned tr,
                                            unsigned stride) {
  unsigned long long ga = (unsigned long long)(size_t)gptr;
  v4u g0;
  g0[0] = 1u;                                  // count=1, no gather
  g0[1] = lds_addr;                            // D# lds_addr
  g0[2] = (unsigned)ga;                        // global_addr[31:0]
  g0[3] = ((unsigned)(ga >> 32) & 0x01FFFFFFu) | 0x80000000u;  // addr hi | type=2
  v8i g1;
  g1[0] = 0x00010000;                          // data_size = 2 bytes
  g1[1] = (int)(tk << 16);                     // tensor_dim0 = tk
  g1[2] = (int)(tr << 16);                     // tensor_dim1 = tr
  g1[3] = (int)(tk << 16);                     // tile_dim0 = tk
  g1[4] = (int)tr;                             // tile_dim1 = tr, tile_dim2 = 0
  g1[5] = (int)stride;                         // tensor_dim0_stride lo32
  g1[6] = 0;
  g1[7] = 0;
  v4i gz4 = {0, 0, 0, 0};
  v8i gz8 = {0, 0, 0, 0, 0, 0, 0, 0};
  __builtin_amdgcn_tensor_load_to_lds(g0, g1, gz4, gz4, gz8, 0);
}

// Gather-mode load: 16 rows of `HH` bf16 each, 16-bit row indices in groups 2/3.
__device__ __forceinline__ void tdm_gather16(unsigned lds_addr, const void* gbase,
                                             const int* idx) {
  unsigned long long ga = (unsigned long long)(size_t)gbase;
  v4u g0;
  g0[0] = 0x80000001u;                         // gather_mode=1, 16-bit idx, count=1
  g0[1] = lds_addr;
  g0[2] = (unsigned)ga;
  g0[3] = ((unsigned)(ga >> 32) & 0x01FFFFFFu) | 0x80000000u;
  v8i g1;
  g1[0] = 0x00010000;                          // data_size = 2 bytes
  g1[1] = (int)(HH << 16);                     // tensor_dim0 = 1024
  g1[2] = (int)(TT << 16);                     // tensor_dim1 = 8192
  g1[3] = (int)(HH << 16);                     // tile_dim0 = 1024
  g1[4] = 16;                                  // tile_dim1 = #indices
  g1[5] = HH;                                  // row stride
  g1[6] = 0;
  g1[7] = 0;
  v4i g2, g3;
  g2[0] = (idx[0]  & 0xFFFF) | (idx[1]  << 16);
  g2[1] = (idx[2]  & 0xFFFF) | (idx[3]  << 16);
  g2[2] = (idx[4]  & 0xFFFF) | (idx[5]  << 16);
  g2[3] = (idx[6]  & 0xFFFF) | (idx[7]  << 16);
  g3[0] = (idx[8]  & 0xFFFF) | (idx[9]  << 16);
  g3[1] = (idx[10] & 0xFFFF) | (idx[11] << 16);
  g3[2] = (idx[12] & 0xFFFF) | (idx[13] << 16);
  g3[3] = (idx[14] & 0xFFFF) | (idx[15] << 16);
  v8i gz8 = {0, 0, 0, 0, 0, 0, 0, 0};
  __builtin_amdgcn_tensor_load_to_lds(g0, g1, g2, g3, gz8, 0);
}

// ------------------------- fp32 -> bf16 conversion -------------------------
__global__ void cvt_f32_bf16_kernel(const float* __restrict__ in,
                                    __bf16* __restrict__ out, size_t n8) {
  size_t i = (size_t)blockIdx.x * blockDim.x + threadIdx.x;
  size_t stride = (size_t)gridDim.x * blockDim.x;
  for (; i < n8; i += stride) {
    float4 a = ((const float4*)in)[2 * i];
    float4 b = ((const float4*)in)[2 * i + 1];
    v8bf o;
    o[0] = (__bf16)a.x; o[1] = (__bf16)a.y; o[2] = (__bf16)a.z; o[3] = (__bf16)a.w;
    o[4] = (__bf16)b.x; o[5] = (__bf16)b.y; o[6] = (__bf16)b.z; o[7] = (__bf16)b.w;
    ((v8bf*)out)[i] = o;
  }
}

// ------------------------------- zero fills --------------------------------
__global__ void zero_f32_kernel(float* __restrict__ p, size_t n) {
  size_t i = (size_t)blockIdx.x * blockDim.x + threadIdx.x;
  size_t stride = (size_t)gridDim.x * blockDim.x;
  for (; i < n; i += stride) p[i] = 0.0f;
}

__global__ void zero_counters_kernel(int* cnt, int* selcnt, float* probsum) {
  int i = threadIdx.x;
  if (i < EE) { cnt[i] = 0; selcnt[i] = 0; probsum[i] = 0.0f; }
}

// ------------------------------ gate / top-2 -------------------------------
__global__ __launch_bounds__(256) void gate_kernel(
    const float* __restrict__ x, const float* __restrict__ gw,
    int* __restrict__ cnt, int* __restrict__ lists, float* __restrict__ combw,
    float* __restrict__ probsum, int* __restrict__ selcnt) {
  const int wave = threadIdx.x >> 5;
  const int lane = threadIdx.x & 31;
  const int t = blockIdx.x * 8 + wave;
  if (t >= TT) return;

  const float* xr = x + (size_t)t * HH;
  float acc[EE];
#pragma unroll
  for (int e = 0; e < EE; ++e) acc[e] = 0.0f;

  for (int i = lane; i < HH; i += 32) {
    float xv = xr[i];
#pragma unroll
    for (int e = 0; e < EE; ++e) acc[e] += xv * gw[e * HH + i];
  }
#pragma unroll
  for (int e = 0; e < EE; ++e) {
    float v = acc[e];
#pragma unroll
    for (int off = 16; off > 0; off >>= 1) v += __shfl_xor(v, off, 32);
    acc[e] = v;
  }

  if (lane == 0) {
    float mx = acc[0];
#pragma unroll
    for (int e = 1; e < EE; ++e) mx = fmaxf(mx, acc[e]);
    float p[EE], s = 0.0f;
#pragma unroll
    for (int e = 0; e < EE; ++e) { p[e] = __expf(acc[e] - mx); s += p[e]; }
    float inv = 1.0f / s;
#pragma unroll
    for (int e = 0; e < EE; ++e) p[e] *= inv;

    int i0 = 0;
#pragma unroll
    for (int e = 1; e < EE; ++e) if (p[e] > p[i0]) i0 = e;
    int i1 = (i0 == 0) ? 1 : 0;
#pragma unroll
    for (int e = 0; e < EE; ++e) if (e != i0 && p[e] > p[i1]) i1 = e;

    float s2 = p[i0] + p[i1] + 1e-6f;
    float w0 = p[i0] / s2;
    float w1 = p[i1] / s2;

#pragma unroll
    for (int e = 0; e < EE; ++e) atomicAdd(&probsum[e], p[e]);
    atomicAdd(&selcnt[i0], 1);
    atomicAdd(&selcnt[i1], 1);

    int p0 = atomicAdd(&cnt[i0], 1);
    lists[i0 * TT + p0] = t;
    combw[i0 * TT + p0] = w0;
    int p1 = atomicAdd(&cnt[i1], 1);
    lists[i1 * TT + p1] = t;
    combw[i1 * TT + p1] = w1;
  }
}

__global__ void aux_kernel(const int* __restrict__ selcnt,
                           const float* __restrict__ probsum,
                           float* __restrict__ out_aux) {
  if (threadIdx.x == 0 && blockIdx.x == 0) {
    float s = 0.0f;
    const float invT = 1.0f / (float)TT;
#pragma unroll
    for (int e = 0; e < EE; ++e)
      s += ((float)selcnt[e] * invT) * (probsum[e] * invT);
    *out_aux = s * (float)EE;
  }
}

// ---------------------------- fused expert FFN -----------------------------
// 256 threads = 8 waves, TM=16 gathered tokens per block.
// TDM streams all matrix operands into LDS:
//   * x tile: one gather-mode TDM (16 token row indices).
//   * w1/w3 (phase1) and w2 (phase2) panels: per-wave double-buffered TDM
//     pipelines into wave-private LDS slices -> no barriers in the k-loop.
// WMMA consumes A/B fragments from LDS; f32 accumulate in VGPRs.
__global__ __launch_bounds__(256, 1) void moe_expert_kernel(
    const __bf16* __restrict__ xbf, const __bf16* __restrict__ w1,
    const __bf16* __restrict__ w2, const __bf16* __restrict__ w3,
    const int* __restrict__ cnt, const int* __restrict__ lists,
    const float* __restrict__ combw, float* __restrict__ out) {
  const int e = blockIdx.y;
  const int c = cnt[e];
  const int base = blockIdx.x * TM;
  if (base >= c) return;

  __shared__ __bf16 xlds[TM * HH];     // 32 KB gathered token tile
  __shared__ __bf16 hlds[TM * ICH];    // 8 KB  hidden chunk
  __shared__ __bf16 wpan[32768];       // 64 KB double-buffered weight panels
  __shared__ int   stk[TM];
  __shared__ float scw[TM];

  const int tid = threadIdx.x;
  const int wid = tid >> 5;
  const int lane = tid & 31;
  const int lh = lane >> 4;
  const int ll = lane & 15;

  if (tid < TM) {
    int idx = base + tid;
    int src = (idx < c) ? idx : (c - 1);
    stk[tid] = lists[e * TT + src];
    scw[tid] = (idx < c) ? combw[e * TT + idx] : 0.0f;
  }

  const size_t wb13 = (size_t)e * II * HH;   // w1/w3 expert base [I,H]
  const size_t wb2  = (size_t)e * HH * II;   // w2 expert base [H,I]

  // x tile via gather-mode TDM (wave 0 issues; indices are uniform scalars).
  if (wid == 0) {
    int idxv[16];
#pragma unroll
    for (int j = 0; j < 16; ++j) {
      int q = base + j; if (q >= c) q = c - 1;
      idxv[j] = lists[e * TT + q];
    }
    tdm_gather16((unsigned)(size_t)xlds, xbf, idxv);
  }

  // phase1: per-wave slice = 32 cols x 32 k per matrix per buffer.
  auto issue_p1 = [&](int ch, int ks, int buf) {
    const size_t col0 = (size_t)(ch * ICH + wid * 32);
    const int k0 = ks * 32;
    tdm_load_2d((unsigned)(size_t)(wpan + buf * 16384 + wid * 1024),
                w1 + wb13 + col0 * HH + k0, 32, 32, HH);
    tdm_load_2d((unsigned)(size_t)(wpan + buf * 16384 + 8192 + wid * 1024),
                w3 + wb13 + col0 * HH + k0, 32, 32, HH);
  };
  // phase2: per-wave slice = 64 cols x 32 k per buffer (2 passes over H).
  auto issue_p2 = [&](int ch, int s, int buf) {
    const size_t col0 = (size_t)((s >> 3) * 512 + wid * 64);
    const int k2 = (s & 7) * 32;
    tdm_load_2d((unsigned)(size_t)(wpan + buf * 16384 + wid * 2048),
                w2 + wb2 + col0 * II + ch * ICH + k2, 32, 64, II);
  };

  issue_p1(0, 0, 0);
  if (wid == 0) __builtin_amdgcn_s_wait_tensorcnt(2);  // gather (issued first) done
  __syncthreads();                                     // xlds + stk visible

  v8f oacc[8];
#pragma unroll
  for (int i = 0; i < 8; ++i) oacc[i] = {};

  for (int ch = 0; ch < II / ICH; ++ch) {
    if (ch > 0) issue_p1(ch, 0, 0);

    v8f h1a[2], h3a[2];
#pragma unroll
    for (int i = 0; i < 2; ++i) { h1a[i] = {}; h3a[i] = {}; }

    auto p1_compute = [&](int ks, int buf) {
      const v16bf a = *(const v16bf*)(xlds + ll * HH + ks * 32 + lh * 16);
      const __bf16* pb = wpan + buf * 16384 + wid * 1024;
#pragma unroll
      for (int t2 = 0; t2 < 2; ++t2) {
        const v16bf b1 = *(const v16bf*)(pb + (t2 * 16 + ll) * 32 + lh * 16);
        const v16bf b3 = *(const v16bf*)(pb + 8192 + (t2 * 16 + ll) * 32 + lh * 16);
        h1a[t2] = wmma_bf16(a, b1, h1a[t2]);
        h3a[t2] = wmma_bf16(a, b3, h3a[t2]);
      }
    };

#pragma unroll
    for (int ks = 0; ks < 31; ++ks) {          // software pipeline, depth 1
      issue_p1(ch, ks + 1, (ks + 1) & 1);
      __builtin_amdgcn_s_wait_tensorcnt(2);    // step ks landed (in-order)
      p1_compute(ks, ks & 1);
    }
    __builtin_amdgcn_s_wait_tensorcnt(0);
    p1_compute(31, 1);

    // hdn = silu(h1) * h3 -> bf16 LDS chunk [16, 256]
#pragma unroll
    for (int t2 = 0; t2 < 2; ++t2) {
#pragma unroll
      for (int j = 0; j < 8; ++j) {
        const int m = j + lh * 8;
        const int nn = wid * 32 + t2 * 16 + ll;
        const float h1v = h1a[t2][j];
        const float sil = h1v / (1.0f + __expf(-h1v));
        hlds[m * ICH + nn] = (__bf16)(sil * h3a[t2][j]);
      }
    }
    __syncthreads();  // hlds ready; all phase1 panel reads done

    issue_p2(ch, 0, 0);
    auto p2_compute = [&](int s, int buf) {
      const int p = s >> 3;
      const int k2 = (s & 7) * 32;
      const v16bf a = *(const v16bf*)(hlds + ll * ICH + k2 + lh * 16);
      const __bf16* pb = wpan + buf * 16384 + wid * 2048;
#pragma unroll
      for (int t2 = 0; t2 < 4; ++t2) {
        const v16bf b = *(const v16bf*)(pb + (t2 * 16 + ll) * 32 + lh * 16);
        oacc[p * 4 + t2] = wmma_bf16(a, b, oacc[p * 4 + t2]);
      }
    };
#pragma unroll
    for (int s = 0; s < 15; ++s) {
      issue_p2(ch, s + 1, (s + 1) & 1);
      __builtin_amdgcn_s_wait_tensorcnt(1);
      p2_compute(s, s & 1);
    }
    __builtin_amdgcn_s_wait_tensorcnt(0);
    p2_compute(15, 1);

    __syncthreads();  // hlds/wpan consumed before next chunk rewrites them
  }

  // ---- epilogue: weighted scatter-add into d_out ----
#pragma unroll
  for (int t2 = 0; t2 < 8; ++t2) {
#pragma unroll
    for (int j = 0; j < 8; ++j) {
      const int m = j + lh * 8;
      if (base + m < c) {
        const int p = t2 >> 2;
        const int nn = p * 512 + wid * 64 + (t2 & 3) * 16 + ll;
        atomicAdd(&out[(size_t)stk[m] * HH + nn], scw[m] * oacc[t2][j]);
      }
    }
  }
}

// ------------------------------ host launcher ------------------------------
extern "C" void kernel_launch(void* const* d_in, const int* in_sizes, int n_in,
                              void* d_out, int out_size, void* d_ws, size_t ws_size,
                              hipStream_t stream) {
  const float* x  = (const float*)d_in[0];   // [T, H]
  const float* gw = (const float*)d_in[1];   // [E, H]
  const float* w1 = (const float*)d_in[2];   // [E, I, H]
  const float* w2 = (const float*)d_in[3];   // [E, H, I]
  const float* w3 = (const float*)d_in[4];   // [E, I, H]
  float* out = (float*)d_out;                // [T*H] + aux

  const size_t nX = (size_t)TT * HH;
  const size_t nW = (size_t)EE * II * HH;

  char* ws = (char*)d_ws;
  size_t off = 0;
  __bf16* xbf = (__bf16*)(ws + off); off += nX * 2;
  __bf16* w1b = (__bf16*)(ws + off); off += nW * 2;
  __bf16* w2b = (__bf16*)(ws + off); off += nW * 2;
  __bf16* w3b = (__bf16*)(ws + off); off += nW * 2;
  int*   cnt     = (int*)(ws + off);   off += 256;
  int*   selcnt  = (int*)(ws + off);   off += 256;
  float* probsum = (float*)(ws + off); off += 256;
  int*   lists   = (int*)(ws + off);   off += (size_t)EE * TT * 4;
  float* combw   = (float*)(ws + off); off += (size_t)EE * TT * 4;

  zero_f32_kernel<<<2048, 256, 0, stream>>>(out, nX + 1);
  zero_counters_kernel<<<1, 64, 0, stream>>>(cnt, selcnt, probsum);

  cvt_f32_bf16_kernel<<<2048, 256, 0, stream>>>(x,  xbf, nX / 8);
  cvt_f32_bf16_kernel<<<4096, 256, 0, stream>>>(w1, w1b, nW / 8);
  cvt_f32_bf16_kernel<<<4096, 256, 0, stream>>>(w2, w2b, nW / 8);
  cvt_f32_bf16_kernel<<<4096, 256, 0, stream>>>(w3, w3b, nW / 8);

  gate_kernel<<<TT / 8, 256, 0, stream>>>(x, gw, cnt, lists, combw, probsum, selcnt);
  aux_kernel<<<1, 32, 0, stream>>>(selcnt, probsum, out + nX);

  dim3 grid(TT / TM, EE);
  moe_expert_kernel<<<grid, 256, 0, stream>>>(xbf, w1b, w2b, w3b,
                                              cnt, lists, combw, out);
}
